// deformable_transformer_layer_45981919871109
// MI455X (gfx1250) — matrix operand order repaired and checked by reading, not compiled, not run on hardware
//
#include <hip/hip_runtime.h>
#include <hip/hip_bf16.h>
#include <math.h>

// ---------------------------------------------------------------------------
// Deformable attention transformer layer for MI455X (gfx1250, wave32, WMMA).
// Dense matmuls: v_wmma_f32_16x16x32_bf16 (f32 accumulate).
// Big GEMMs: LDS tiles staged with global_load_async_to_lds_b128 (ASYNCcnt).
// ---------------------------------------------------------------------------

typedef __attribute__((ext_vector_type(16))) __bf16 v16bf;
typedef __attribute__((ext_vector_type(8)))  float  v8f;

#define LANE() (threadIdx.x & 31)

// D = A(16x32 bf16) * B(32x16 bf16) + C  (f32 accum)
__device__ __forceinline__ v8f wmma_bf16(v16bf a, v16bf b, v8f c) {
  return __builtin_amdgcn_wmma_f32_16x16x32_bf16(false, a, false, b,
                                                 (short)0, c, false, false);
}

// K index inside a 16x32 A fragment for element e (contiguous: k = 16*half + e, matches B loaders).
__device__ __forceinline__ int a_frag_k(int e, int half) {
  return e + (half << 4);
}

// A fragment from f32 source; element (m,k) at src[m*rs + k*cs], scaled by mul.
__device__ __forceinline__ v16bf load_a_f32(const float* __restrict__ src,
                                            int rs, int cs, float mul) {
  const int l = LANE();
  const int m = l & 15, half = l >> 4;
  v16bf a;
#pragma unroll
  for (int e = 0; e < 16; ++e) {
    int k = a_frag_k(e, half);
    a[e] = (__bf16)(src[(long)m * rs + (long)k * cs] * mul);
  }
  return a;
}

// B fragment from f32 source; element (k,n) at src[k*rs + n*cs].
__device__ __forceinline__ v16bf load_b_f32(const float* __restrict__ src,
                                            int rs, int cs) {
  const int l = LANE();
  const int n = l & 15;
  const int k0 = (l >> 4) << 4;
  v16bf b;
#pragma unroll
  for (int e = 0; e < 16; ++e)
    b[e] = (__bf16)src[(long)(k0 + e) * rs + (long)n * cs];
  return b;
}

// Fragment loaders from bf16 LDS (row-major, leading dim ld).
__device__ __forceinline__ v16bf load_a_ldsb(const __bf16* src, int ld) {
  const int l = LANE();
  const int m = l & 15, half = l >> 4;
  v16bf a;
#pragma unroll
  for (int e = 0; e < 16; ++e) a[e] = src[m * ld + a_frag_k(e, half)];
  return a;
}
__device__ __forceinline__ v16bf load_b_ldsb(const __bf16* src, int ld) {
  const int l = LANE();
  const int n = l & 15;
  const int k0 = (l >> 4) << 4;
  v16bf b;
#pragma unroll
  for (int e = 0; e < 16; ++e) b[e] = src[(k0 + e) * ld + n];
  return b;
}

// Fragment loaders from f32 LDS tiles (convert to bf16 on read).
__device__ __forceinline__ v16bf load_a_ldsf(const float* src, int ld) {
  const int l = LANE();
  const int m = l & 15, half = l >> 4;
  v16bf a;
#pragma unroll
  for (int e = 0; e < 16; ++e) a[e] = (__bf16)src[m * ld + a_frag_k(e, half)];
  return a;
}
__device__ __forceinline__ v16bf load_b_ldsf(const float* src, int ld) {
  const int l = LANE();
  const int n = l & 15;
  const int k0 = (l >> 4) << 4;
  v16bf b;
#pragma unroll
  for (int e = 0; e < 16; ++e) b[e] = (__bf16)src[(k0 + e) * ld + n];
  return b;
}

// gfx1250 async global->LDS 16-byte copy (ASYNCcnt tracked).
// LDS aperture addresses keep the DS offset in bits [31:0] (ISA 10.2).
__device__ __forceinline__ void async_copy_b128(const float* gptr, float* lptr) {
  unsigned long long ga = (unsigned long long)gptr;
  unsigned lds = (unsigned)(unsigned long long)lptr;
  asm volatile("global_load_async_to_lds_b128 %0, %1, off"
               :: "v"(lds), "v"(ga) : "memory");
}
__device__ __forceinline__ void async_wait0() {
  asm volatile("s_wait_asynccnt 0x0" ::: "memory");
}

__device__ __forceinline__ float gelu_exact(float x) {
  return 0.5f * x * (1.0f + erff(x * 0.70710678118654752f));
}

// ---------------------------------------------------------------------------
// K1: eval-mode BatchNorm2d (running stats 0/1):  y = x * g/sqrt(1+eps) + b
// ---------------------------------------------------------------------------
__global__ void bn_kernel(const float* __restrict__ x, const float* __restrict__ g,
                          const float* __restrict__ bta, float* __restrict__ y,
                          int total) {
  int idx = blockIdx.x * blockDim.x + threadIdx.x;
  if (idx >= total) return;
  int c = (idx >> 10) & 511;
  const float s = rsqrtf(1.0f + 1e-5f);
  y[idx] = x[idx] * (g[c] * s) + bta[c];
}

// ---------------------------------------------------------------------------
// K2/K6: grouped 1x1 conv via WMMA.  Per (b,g): Y(64 x npos) = W(64x64) X(64 x npos)
// ---------------------------------------------------------------------------
__global__ void grouped_proj_kernel(const float* __restrict__ X,
                                    const float* __restrict__ Wg,
                                    float* __restrict__ Y, int npos) {
  const int wave = threadIdx.x >> 5;
  const int n0 = (blockIdx.x * 4 + wave) * 16;
  if (n0 >= npos) return;
  const int m0 = blockIdx.y * 16;
  const int bg = blockIdx.z;
  const int b = bg >> 3, g = bg & 7;
  const float* Wb = Wg + g * 4096;
  const float* Xb = X + ((long)b * 512 + g * 64) * npos;
  float* Yb = Y + ((long)b * 512 + g * 64) * npos;

  v8f acc = {};
  acc = wmma_bf16(load_a_f32(Wb + m0 * 64, 64, 1, 1.0f),
                  load_b_f32(Xb + n0, npos, 1), acc);
  acc = wmma_bf16(load_a_f32(Wb + m0 * 64 + 32, 64, 1, 1.0f),
                  load_b_f32(Xb + (long)32 * npos + n0, npos, 1), acc);

  const int col = LANE() & 15, half = LANE() >> 4;
#pragma unroll
  for (int r = 0; r < 8; ++r)
    Yb[(long)(m0 + half * 8 + r) * npos + n0 + col] = acc[r];
}

// ---------------------------------------------------------------------------
// K3: depthwise 6x6 conv stride 4 pad 1 + GELU.  (16,64,32,32) -> (16,64,8,8)
// ---------------------------------------------------------------------------
__global__ void off_dw_kernel(const float* __restrict__ q,
                              const float* __restrict__ dw,
                              const float* __restrict__ dwb,
                              float* __restrict__ mid) {
  int idx = blockIdx.x * blockDim.x + threadIdx.x;   // 65536
  if (idx >= 65536) return;
  int p = idx & 63, c = (idx >> 6) & 63, bg = idx >> 12;
  int od = p >> 3, ow = p & 7;
  int b = bg >> 3, g = bg & 7;
  const float* img = q + ((long)b * 512 + g * 64 + c) * 1024;
  float s = 0.0f;
#pragma unroll
  for (int ky = 0; ky < 6; ++ky) {
    int iy = od * 4 - 1 + ky;
    if (iy < 0 || iy > 31) continue;
#pragma unroll
    for (int kx = 0; kx < 6; ++kx) {
      int ix = ow * 4 - 1 + kx;
      if (ix < 0 || ix > 31) continue;
      s += img[iy * 32 + ix] * dw[c * 36 + ky * 6 + kx];
    }
  }
  mid[idx] = gelu_exact(s + dwb[c]);
}

// ---------------------------------------------------------------------------
// K4: 1x1 64->2, tanh*4, add base grid, normalize -> vgrid_scaled (nx,ny)
// ---------------------------------------------------------------------------
__global__ void off_pw_kernel(const float* __restrict__ mid,
                              const float* __restrict__ pw,
                              float* __restrict__ gridb) {
  int idx = blockIdx.x * blockDim.x + threadIdx.x;   // 1024
  if (idx >= 1024) return;
  int p = idx & 63, bg = idx >> 6;
  float ox = 0.0f, oy = 0.0f;
  for (int c = 0; c < 64; ++c) {
    float m = mid[bg * 4096 + c * 64 + p];
    ox += m * pw[c];
    oy += m * pw[64 + c];
  }
  ox = tanhf(ox) * 4.0f;
  oy = tanhf(oy) * 4.0f;
  int od = p >> 3, ow = p & 7;
  float vx = (float)ow + ox;
  float vy = (float)od + oy;
  gridb[bg * 128 + p * 2 + 0] = 2.0f * vx / 7.0f - 1.0f;
  gridb[bg * 128 + p * 2 + 1] = 2.0f * vy / 7.0f - 1.0f;
}

// ---------------------------------------------------------------------------
// K5: bilinear grid_sample (zeros padding, align_corners=False).
// ---------------------------------------------------------------------------
__global__ void grid_sample_kernel(const float* __restrict__ xn,
                                   const float* __restrict__ gridb,
                                   float* __restrict__ kv) {
  int idx = blockIdx.x * blockDim.x + threadIdx.x;   // 65536
  if (idx >= 65536) return;
  int p = idx & 63, c = (idx >> 6) & 63, bg = idx >> 12;
  int b = bg >> 3, g = bg & 7;
  float nx = gridb[bg * 128 + p * 2 + 0];
  float ny = gridb[bg * 128 + p * 2 + 1];
  float gx = (nx + 1.0f) * 16.0f - 0.5f;
  float gy = (ny + 1.0f) * 16.0f - 0.5f;
  float x0f = floorf(gx), y0f = floorf(gy);
  float wx = gx - x0f, wy = gy - y0f;
  int x0 = (int)x0f, y0 = (int)y0f;
  const float* img = xn + ((long)b * 512 + g * 64 + c) * 1024;
  auto tap = [&](int xx, int yy) -> float {
    if (xx < 0 || xx > 31 || yy < 0 || yy > 31) return 0.0f;
    return img[yy * 32 + xx];
  };
  float v = tap(x0, y0) * (1.0f - wx) * (1.0f - wy)
          + tap(x0 + 1, y0) * wx * (1.0f - wy)
          + tap(x0, y0 + 1) * (1.0f - wx) * wy
          + tap(x0 + 1, y0 + 1) * wx * wy;
  kv[(long)b * 32768 + (g * 64 + c) * 64 + p] = v;
}

// ---------------------------------------------------------------------------
// K7: CPB bias MLP (2->128->128->1) over 1,048,576 (bg,i,j) rows. WMMA core.
// ---------------------------------------------------------------------------
__global__ __launch_bounds__(128)
void cpb_bias_kernel(const float* __restrict__ gridb,
                     const float* __restrict__ w1, const float* __restrict__ b1,
                     const float* __restrict__ w2, const float* __restrict__ b2,
                     const float* __restrict__ w3, const float* __restrict__ b3,
                     float* __restrict__ biasb) {
  __shared__ __bf16 sW2[128 * 128];        // 32 KB
  __shared__ __bf16 sH1[4][16 * 128];      // 16 KB
  const int tid = threadIdx.x;             // 128 threads, 4 waves
  for (int t = tid; t < 128 * 128; t += 128) sW2[t] = (__bf16)w2[t];
  __syncthreads();

  const int wave = tid >> 5;
  const int lane = tid & 31;
  const int tile = blockIdx.x * 4 + wave;         // 65536 tiles total
  const long r0 = (long)tile * 16;
  const int bg = (int)(r0 >> 16);
  const int i  = (int)((r0 & 65535) >> 6);
  const int j0 = (int)(r0 & 63);

  const float gq0 = 2.0f * (float)(i & 31) / 31.0f - 1.0f;
  const float gq1 = 2.0f * (float)(i >> 5) / 31.0f - 1.0f;

  // layer 1 (each lane: one row, one 64-col half)
  const int m = lane & 15;
  const int chalf = (lane >> 4) * 64;
  const int j = j0 + m;
  const float nx = gridb[bg * 128 + j * 2 + 0];
  const float ny = gridb[bg * 128 + j * 2 + 1];
  float p0 = gq0 - nx, p1 = gq1 - ny;
  float t0 = copysignf(log1pf(fabsf(p0)), p0);
  float t1 = copysignf(log1pf(fabsf(p1)), p1);
  __bf16* h1row = &sH1[wave][m * 128];
  for (int c = 0; c < 64; ++c) {
    int cc = chalf + c;
    float h = fmaf(t0, w1[cc], fmaf(t1, w1[128 + cc], b1[cc]));
    h1row[cc] = (__bf16)fmaxf(h, 0.0f);
  }
  __syncthreads();

  // layer 2 (WMMA) + fused layer 3
  const int col = lane & 15, half = lane >> 4;
  float dot[8] = {0, 0, 0, 0, 0, 0, 0, 0};
#pragma unroll
  for (int nt = 0; nt < 8; ++nt) {
    v8f acc = {};
#pragma unroll
    for (int ks = 0; ks < 4; ++ks)
      acc = wmma_bf16(load_a_ldsb(&sH1[wave][ks * 32], 128),
                      load_b_ldsb(&sW2[(ks * 32) * 128 + nt * 16], 128), acc);
    int o = nt * 16 + col;
    float b2o = b2[o], w3o = w3[o];
#pragma unroll
    for (int r = 0; r < 8; ++r)
      dot[r] += fmaxf(acc[r] + b2o, 0.0f) * w3o;
  }
#pragma unroll
  for (int r = 0; r < 8; ++r) {
    float v = dot[r];
    v += __shfl_xor(v, 1, 32);
    v += __shfl_xor(v, 2, 32);
    v += __shfl_xor(v, 4, 32);
    v += __shfl_xor(v, 8, 32);
    dot[r] = v;
  }
  if (col == 0) {
    const float bb3 = b3[0];
#pragma unroll
    for (int r = 0; r < 8; ++r) {
      int row = half * 8 + r;
      biasb[(long)bg * 65536 + (long)i * 64 + (j0 + row)] = dot[r] + bb3;
    }
  }
}

// ---------------------------------------------------------------------------
// K8a: sim = (q^T * scale) k + bias.   Per (b,h): M=1024(i), N=64(j), K=64(d)
// ---------------------------------------------------------------------------
__global__ void attn_sim_kernel(const float* __restrict__ q,
                                const float* __restrict__ kb,
                                const float* __restrict__ biasb,
                                float* __restrict__ sim) {
  const int wave = threadIdx.x >> 5;           // n-tile 0..3
  const int i0 = blockIdx.y * 16;
  const int bh = blockIdx.z;
  const int b = bh >> 3, h = bh & 7;
  const float scale = 0.125f;                  // 64^-0.5
  const float* qbase = q + ((long)b * 512 + h * 64) * 1024 + i0;
  const float* kbase = kb + ((long)b * 512 + h * 64) * 64 + wave * 16;

  v8f acc = {};
  acc = wmma_bf16(load_a_f32(qbase, 1, 1024, scale), load_b_f32(kbase, 64, 1), acc);
  acc = wmma_bf16(load_a_f32(qbase + (long)32 * 1024, 1, 1024, scale),
                  load_b_f32(kbase + 32 * 64, 64, 1), acc);

  const int col = LANE() & 15, half = LANE() >> 4;
#pragma unroll
  for (int r = 0; r < 8; ++r) {
    int i = i0 + half * 8 + r;
    int jj = wave * 16 + col;
    sim[((long)bh * 1024 + i) * 64 + jj] =
        acc[r] + biasb[(long)bh * 65536 + (long)i * 64 + jj];
  }
}

// K8b: softmax over j=64, one thread per (b,h,i) row, in place.
__global__ void softmax_kernel(float* __restrict__ sim) {
  int row = blockIdx.x * blockDim.x + threadIdx.x;
  if (row >= 16384) return;
  float* p = sim + (long)row * 64;
  float mx = -3.4e38f;
  for (int jj = 0; jj < 64; ++jj) mx = fmaxf(mx, p[jj]);
  float s = 0.0f;
  for (int jj = 0; jj < 64; ++jj) { float e = __expf(p[jj] - mx); p[jj] = e; s += e; }
  float inv = 1.0f / s;
  for (int jj = 0; jj < 64; ++jj) p[jj] *= inv;
}

// K8c: out = attn @ v^T.  Per (b,h): M=1024(i), N=64(d), K=64(j).
__global__ void attn_out_kernel(const float* __restrict__ attn,
                                const float* __restrict__ vb,
                                float* __restrict__ aout) {
  const int wave = threadIdx.x >> 5;           // d-tile 0..3
  const int i0 = blockIdx.y * 16;
  const int bh = blockIdx.z;
  const int b = bh >> 3, h = bh & 7;
  const float* Ab = attn + ((long)bh * 1024 + i0) * 64;
  const float* Bb = vb + ((long)b * 512 + h * 64 + wave * 16) * 64;

  v8f acc = {};
  acc = wmma_bf16(load_a_f32(Ab, 64, 1, 1.0f), load_b_f32(Bb, 1, 64), acc);
  acc = wmma_bf16(load_a_f32(Ab + 32, 64, 1, 1.0f), load_b_f32(Bb + 32, 1, 64), acc);

  const int col = LANE() & 15, half = LANE() >> 4;
#pragma unroll
  for (int r = 0; r < 8; ++r) {
    int i = i0 + half * 8 + r;
    int d = wave * 16 + col;
    aout[((long)b * 512 + h * 64 + d) * 1024 + i] = acc[r];
  }
}

// ---------------------------------------------------------------------------
// K9/K11/K12: blocked bf16-WMMA GEMM with async global->LDS tile staging.
//   Y[b,m,n] = epi( W[m,:] . X[b,:,n] + bias[m] )
// Block = 256 threads (8 waves) computing a 64x64 tile; per K-step stage
// A (64x32 f32) and B (32x64 f32) via global_load_async_to_lds_b128.
// Wave w: m-tile = (w&3), n-tiles {(w>>2)*2, (w>>2)*2+1} -> A frag reused.
// EPI: 0 = bias, 1 = GELU(.+bias), 2 = bias + residual
// ---------------------------------------------------------------------------
template <int EPI>
__global__ __launch_bounds__(256)
void gemm_bias_kernel(const float* __restrict__ W,
                      const float* __restrict__ X,
                      const float* __restrict__ bias,
                      const float* __restrict__ res,
                      float* __restrict__ Y,
                      int M, int N, int K) {
  __shared__ float sA[64 * 32];   // 8 KB, (m,k) row-major ld=32
  __shared__ float sB[32 * 64];   // 8 KB, (k,n) row-major ld=64
  const int tid = threadIdx.x;
  const int wave = tid >> 5;
  const int m0 = blockIdx.y * 64;
  const int n0 = blockIdx.x * 64;
  const int bb = blockIdx.z;
  const float* Xb = X + (long)bb * K * N;

  const int mt = (wave & 3) * 16;        // wave's m-tile offset in block
  const int ntb = (wave >> 2) * 32;      // wave's n-pair base in block
  v8f acc0 = {}, acc1 = {};

  for (int k0 = 0; k0 < K; k0 += 32) {
    __syncthreads();   // previous-iteration LDS reads done
    // A tile: 64 rows x 32 floats = 512 chunks of 16B
#pragma unroll
    for (int it = 0; it < 2; ++it) {
      int idx = tid + it * 256;
      int row = idx >> 3, ch = (idx & 7) * 4;
      async_copy_b128(W + (long)(m0 + row) * K + k0 + ch, &sA[row * 32 + ch]);
    }
    // B tile: 32 rows x 64 floats = 512 chunks of 16B
#pragma unroll
    for (int it = 0; it < 2; ++it) {
      int idx = tid + it * 256;
      int row = idx >> 4, ch = (idx & 15) * 4;
      async_copy_b128(Xb + (long)(k0 + row) * N + n0 + ch, &sB[row * 64 + ch]);
    }
    async_wait0();
    __syncthreads();   // all async tiles visible to all waves

    v16bf a = load_a_ldsf(&sA[mt * 32], 32);
    acc0 = wmma_bf16(a, load_b_ldsf(&sB[ntb], 64), acc0);
    acc1 = wmma_bf16(a, load_b_ldsf(&sB[ntb + 16], 64), acc1);
  }

  const int col = LANE() & 15, half = LANE() >> 4;
#pragma unroll
  for (int r = 0; r < 8; ++r) {
    int mrow = m0 + mt + half * 8 + r;
    float bs = bias[mrow];
    long o0 = (long)bb * M * N + (long)mrow * N + n0 + ntb + col;
    float v0 = acc0[r] + bs;
    float v1 = acc1[r] + bs;
    if constexpr (EPI == 1) { v0 = gelu_exact(v0); v1 = gelu_exact(v1); }
    if constexpr (EPI == 2) { v0 += res[o0]; v1 += res[o0 + 16]; }
    Y[o0] = v0;
    Y[o0 + 16] = v1;
  }
}

// ---------------------------------------------------------------------------
// launch
// ---------------------------------------------------------------------------
extern "C" void kernel_launch(void* const* d_in, const int* in_sizes, int n_in,
                              void* d_out, int out_size, void* d_ws, size_t ws_size,
                              hipStream_t stream) {
  const float* x        = (const float*)d_in[0];
  const float* bn1_g    = (const float*)d_in[1];
  const float* bn1_b    = (const float*)d_in[2];
  const float* bn2_g    = (const float*)d_in[3];
  const float* bn2_b    = (const float*)d_in[4];
  const float* qw       = (const float*)d_in[5];
  const float* kw       = (const float*)d_in[6];
  const float* vw       = (const float*)d_in[7];
  const float* out_w    = (const float*)d_in[8];
  const float* out_b    = (const float*)d_in[9];
  const float* off_dw_w = (const float*)d_in[10];
  const float* off_dw_b = (const float*)d_in[11];
  const float* off_pw_w = (const float*)d_in[12];
  const float* cpb_w1   = (const float*)d_in[13];
  const float* cpb_b1   = (const float*)d_in[14];
  const float* cpb_w2   = (const float*)d_in[15];
  const float* cpb_b2   = (const float*)d_in[16];
  const float* cpb_w3   = (const float*)d_in[17];
  const float* cpb_b3   = (const float*)d_in[18];
  const float* mlp_w1   = (const float*)d_in[19];
  const float* mlp_b1   = (const float*)d_in[20];
  const float* mlp_w2   = (const float*)d_in[21];
  const float* mlp_b2   = (const float*)d_in[22];
  float* out = (float*)d_out;

  // workspace carve-up (floats)
  float* ws   = (float*)d_ws;
  float* XN   = ws;                 // 1048576  (B,512,1024)
  float* Q    = XN   + 1048576;     // 1048576
  float* MID  = Q    + 1048576;     // 65536    (16,64,64)
  float* GRB  = MID  + 65536;       // 2048     (16,64,2)
  float* KV   = GRB  + 2048;        // 65536    (B,512,64)
  float* KB   = KV   + 65536;       // 65536
  float* VB   = KB   + 65536;       // 65536
  float* BIAS = VB   + 65536;       // 1048576  (16,1024,64)
  float* SIM  = BIAS + 1048576;     // 1048576  (B,8,1024,64)
  float* AOUT = SIM  + 1048576;     // 1048576  (B,512,1024)
  float* X1   = AOUT + 1048576;     // 1048576
  float* XN2  = X1   + 1048576;     // 1048576
  float* H1   = XN2  + 1048576;     // 4194304  (B,2048,1024)

  const int TOT = 2 * 512 * 1024;

  // ---- deformable attention ----
  bn_kernel<<<(TOT + 255) / 256, 256, 0, stream>>>(x, bn1_g, bn1_b, XN, TOT);
  grouped_proj_kernel<<<dim3(16, 4, 16), 128, 0, stream>>>(XN, qw, Q, 1024);
  off_dw_kernel<<<256, 256, 0, stream>>>(Q, off_dw_w, off_dw_b, MID);
  off_pw_kernel<<<4, 256, 0, stream>>>(MID, off_pw_w, GRB);
  grid_sample_kernel<<<256, 256, 0, stream>>>(XN, GRB, KV);
  grouped_proj_kernel<<<dim3(1, 4, 16), 128, 0, stream>>>(KV, kw, KB, 64);
  grouped_proj_kernel<<<dim3(1, 4, 16), 128, 0, stream>>>(KV, vw, VB, 64);
  cpb_bias_kernel<<<16384, 128, 0, stream>>>(GRB, cpb_w1, cpb_b1, cpb_w2, cpb_b2,
                                             cpb_w3, cpb_b3, BIAS);
  attn_sim_kernel<<<dim3(1, 64, 16), 128, 0, stream>>>(Q, KB, BIAS, SIM);
  softmax_kernel<<<64, 256, 0, stream>>>(SIM);
  attn_out_kernel<<<dim3(1, 64, 16), 128, 0, stream>>>(SIM, VB, AOUT);
  // out projection + residual
  gemm_bias_kernel<2><<<dim3(16, 8, 2), 256, 0, stream>>>(out_w, AOUT, out_b, x, X1,
                                                          512, 1024, 512);
  // ---- conv1x1 MLP ----
  bn_kernel<<<(TOT + 255) / 256, 256, 0, stream>>>(X1, bn2_g, bn2_b, XN2, TOT);
  gemm_bias_kernel<1><<<dim3(16, 32, 2), 256, 0, stream>>>(mlp_w1, XN2, mlp_b1, nullptr,
                                                           H1, 2048, 1024, 512);
  gemm_bias_kernel<2><<<dim3(16, 8, 2), 256, 0, stream>>>(mlp_w2, H1, mlp_b2, X1, out,
                                                          512, 1024, 2048);
}
